// GRNNRegressor_23106924052821
// MI455X (gfx1250) — compile-verified, matrix operand order checked by reading, not run on hardware
//
#include <hip/hip_runtime.h>

typedef __attribute__((ext_vector_type(16))) _Float16 v16h;
typedef __attribute__((ext_vector_type(8)))  _Float16 v8h;
typedef __attribute__((ext_vector_type(8)))  float    v8f;

// exp(-d/2) = exp2(-K*d),  K = 1/(2 ln 2)
#define K_HALF_LOG2E 0.72134752044448170f
#define TWO_K        1.4426950408889634f

// ---------------------------------------------------------------------------
// Fragment loader from fp32 rows (A matrix, and B in the fallback path).
// CDNA5 16-bit A/B layout: lane (lo, hi) holds, per 32-K fragment f:
//   elements [0..7]  = K  f*32 + hi*8      .. +7
//   elements [8..15] = K  f*32 + 16 + hi*8 .. +7
// Returns this lane's partial sum of squares (32 of 64 K values).
// ---------------------------------------------------------------------------
__device__ __forceinline__ float load_frags_f32(const float* __restrict__ row, int hi,
                                                v16h& f0, v16h& f1) {
    const float4* r4 = reinterpret_cast<const float4*>(row);
    const int b = hi * 2;
    float4 q0 = r4[b + 0],  q1 = r4[b + 1];
    float4 q2 = r4[b + 4],  q3 = r4[b + 5];
    float4 q4 = r4[b + 8],  q5 = r4[b + 9];
    float4 q6 = r4[b + 12], q7 = r4[b + 13];

    f0[0]=(_Float16)q0.x;  f0[1]=(_Float16)q0.y;  f0[2]=(_Float16)q0.z;  f0[3]=(_Float16)q0.w;
    f0[4]=(_Float16)q1.x;  f0[5]=(_Float16)q1.y;  f0[6]=(_Float16)q1.z;  f0[7]=(_Float16)q1.w;
    f0[8]=(_Float16)q2.x;  f0[9]=(_Float16)q2.y;  f0[10]=(_Float16)q2.z; f0[11]=(_Float16)q2.w;
    f0[12]=(_Float16)q3.x; f0[13]=(_Float16)q3.y; f0[14]=(_Float16)q3.z; f0[15]=(_Float16)q3.w;
    f1[0]=(_Float16)q4.x;  f1[1]=(_Float16)q4.y;  f1[2]=(_Float16)q4.z;  f1[3]=(_Float16)q4.w;
    f1[4]=(_Float16)q5.x;  f1[5]=(_Float16)q5.y;  f1[6]=(_Float16)q5.z;  f1[7]=(_Float16)q5.w;
    f1[8]=(_Float16)q6.x;  f1[9]=(_Float16)q6.y;  f1[10]=(_Float16)q6.z; f1[11]=(_Float16)q6.w;
    f1[12]=(_Float16)q7.x; f1[13]=(_Float16)q7.y; f1[14]=(_Float16)q7.z; f1[15]=(_Float16)q7.w;

    float s = 0.0f;
    s += q0.x*q0.x + q0.y*q0.y + q0.z*q0.z + q0.w*q0.w;
    s += q1.x*q1.x + q1.y*q1.y + q1.z*q1.z + q1.w*q1.w;
    s += q2.x*q2.x + q2.y*q2.y + q2.z*q2.z + q2.w*q2.w;
    s += q3.x*q3.x + q3.y*q3.y + q3.z*q3.z + q3.w*q3.w;
    s += q4.x*q4.x + q4.y*q4.y + q4.z*q4.z + q4.w*q4.w;
    s += q5.x*q5.x + q5.y*q5.y + q5.z*q5.z + q5.w*q5.w;
    s += q6.x*q6.x + q6.y*q6.y + q6.z*q6.z + q6.w*q6.w;
    s += q7.x*q7.x + q7.y*q7.y + q7.z*q7.z + q7.w*q7.w;
    return s;
}

// ---------------------------------------------------------------------------
// Prep: X_train fp32 [N,64] -> f16 [N,64] in ws, plus interleaved
// float2 { -K*||t||^2 , y } table (one b64 load in the hot loop).
// ---------------------------------------------------------------------------
__global__ __launch_bounds__(256)
void grnn_prep_kernel(const float* __restrict__ Xt,
                      const float* __restrict__ y,
                      _Float16* __restrict__ Xh,
                      float2* __restrict__ ty,
                      int N) {
    const int n = blockIdx.x * blockDim.x + threadIdx.x;
    if (n >= N) return;
    const float4* src = reinterpret_cast<const float4*>(Xt + (size_t)n * 64);
    v8h* dst = reinterpret_cast<v8h*>(Xh + (size_t)n * 64);
    float s = 0.0f;
#pragma unroll
    for (int i = 0; i < 8; ++i) {
        float4 a = src[2 * i];
        float4 b = src[2 * i + 1];
        s += a.x*a.x + a.y*a.y + a.z*a.z + a.w*a.w;
        s += b.x*b.x + b.y*b.y + b.z*b.z + b.w*b.w;
        v8h h;
        h[0]=(_Float16)a.x; h[1]=(_Float16)a.y; h[2]=(_Float16)a.z; h[3]=(_Float16)a.w;
        h[4]=(_Float16)b.x; h[5]=(_Float16)b.y; h[6]=(_Float16)b.z; h[7]=(_Float16)b.w;
        dst[i] = h;
    }
    ty[n] = make_float2(s * -K_HALF_LOG2E, y[n]);
}

// ---------------------------------------------------------------------------
// Shared epilogue: lane-group + cross-wave reductions, final divide.
// ---------------------------------------------------------------------------
__device__ __forceinline__ void grnn_reduce_store(float num[8], float den[8],
                                                  float (*s_num)[16], float (*s_den)[16],
                                                  int tid, int wave, int lo, int hi,
                                                  int mBase, int B,
                                                  float* __restrict__ out) {
#pragma unroll
    for (int r = 0; r < 8; ++r) {
#pragma unroll
        for (int m = 1; m <= 8; m <<= 1) {
            num[r] += __shfl_xor(num[r], m);
            den[r] += __shfl_xor(den[r], m);
        }
    }
    if (lo == 0) {
#pragma unroll
        for (int r = 0; r < 8; ++r) {
            s_num[wave][hi * 8 + r] = num[r];
            s_den[wave][hi * 8 + r] = den[r];
        }
    }
    __syncthreads();
    if (tid < 16) {
        float n = 0.0f, d = 0.0f;
#pragma unroll
        for (int w = 0; w < 8; ++w) { n += s_num[w][tid]; d += s_den[w][tid]; }
        const int m = mBase + tid;
        if (m < B) out[m] = n / d;
    }
}

// ---------------------------------------------------------------------------
// Fast path. Per full tile: 4x b128 (f16 B frags) + 1x b64 (tsq,y),
// 2x v_wmma, then per element: add + fma + v_exp_f32 + fmac + add.
// No masking, no clamps, no LDS in the hot loop.
// ---------------------------------------------------------------------------
__global__ __launch_bounds__(256)
void grnn_wmma_fast_kernel(const float* __restrict__ x,
                           const _Float16* __restrict__ Xh,
                           const float2* __restrict__ ty_arr,
                           float* __restrict__ out,
                           int B, int N) {
    __shared__ float s_num[8][16];
    __shared__ float s_den[8][16];

    const int tid  = threadIdx.x;
    const int lane = tid & 31;
    const int wave = tid >> 5;
    const int lo   = lane & 15;
    const int hi   = lane >> 4;
    const int mBase = blockIdx.x * 16;

    // A fragments: this wave's 16 query rows (fp32 -> f16 once).
    int arow = mBase + lo;
    if (arow >= B) arow = B - 1;
    v16h a0, a1;
    float xs_part = load_frags_f32(x + (size_t)arow * 64, hi, a0, a1);
    float xsq = xs_part + __shfl_xor(xs_part, 16);

    // C/D-layout rows for this lane, pre-scaled by -K.
    float xs2[8];
#pragma unroll
    for (int r = 0; r < 8; ++r)
        xs2[r] = __shfl(xsq, hi * 8 + r) * -K_HALF_LOG2E;

    float num[8], den[8];
#pragma unroll
    for (int r = 0; r < 8; ++r) { num[r] = 0.0f; den[r] = 0.0f; }

    const v8h* pbase = reinterpret_cast<const v8h*>(Xh);
    const int fullTiles = N >> 4;

#pragma unroll 2
    for (int ct = wave; ct < fullTiles; ct += 8) {
        const int col = ct * 16 + lo;
        const v8h* p = pbase + (size_t)col * 8;
        v8h r0 = p[hi];        // K = hi*8 .. +7
        v8h r1 = p[2 + hi];    // K = 16+hi*8
        v8h r2 = p[4 + hi];    // K = 32+hi*8
        v8h r3 = p[6 + hi];    // K = 48+hi*8
        v16h b0 = __builtin_shufflevector(r0, r1, 0,1,2,3,4,5,6,7,8,9,10,11,12,13,14,15);
        v16h b1 = __builtin_shufflevector(r2, r3, 0,1,2,3,4,5,6,7,8,9,10,11,12,13,14,15);

        const float2 ty = ty_arr[col];      // { -K*t^2 , y }

        v8f c = {0.f, 0.f, 0.f, 0.f, 0.f, 0.f, 0.f, 0.f};
        c = __builtin_amdgcn_wmma_f32_16x16x32_f16(false, a0, false, b0,
                                                   (short)0, c, false, false);
        c = __builtin_amdgcn_wmma_f32_16x16x32_f16(false, a1, false, b1,
                                                   (short)0, c, false, false);

#pragma unroll
        for (int r = 0; r < 8; ++r) {
            // exp2(-K*x^2 - K*t^2 + 2K*cross) == exp(-dist/2)
            float w = __builtin_amdgcn_exp2f(fmaf(c[r], TWO_K, xs2[r] + ty.x));
            den[r] += w;
            num[r] += w * ty.y;
        }
    }

    // Partial tail tile (at most one; owned by a single wave -> EXEC all-ones).
    if ((N & 15) && ((fullTiles & 7) == wave)) {
        const int col  = fullTiles * 16 + lo;
        const int colc = (col < N) ? col : (N - 1);
        const v8h* p = pbase + (size_t)colc * 8;
        v8h r0 = p[hi], r1 = p[2 + hi], r2 = p[4 + hi], r3 = p[6 + hi];
        v16h b0 = __builtin_shufflevector(r0, r1, 0,1,2,3,4,5,6,7,8,9,10,11,12,13,14,15);
        v16h b1 = __builtin_shufflevector(r2, r3, 0,1,2,3,4,5,6,7,8,9,10,11,12,13,14,15);
        const float2 ty = ty_arr[colc];

        v8f c = {0.f, 0.f, 0.f, 0.f, 0.f, 0.f, 0.f, 0.f};
        c = __builtin_amdgcn_wmma_f32_16x16x32_f16(false, a0, false, b0,
                                                   (short)0, c, false, false);
        c = __builtin_amdgcn_wmma_f32_16x16x32_f16(false, a1, false, b1,
                                                   (short)0, c, false, false);

        const float colmask = (col < N) ? 1.0f : 0.0f;
#pragma unroll
        for (int r = 0; r < 8; ++r) {
            float w = __builtin_amdgcn_exp2f(fmaf(c[r], TWO_K, xs2[r] + ty.x)) * colmask;
            den[r] += w;
            num[r] += w * ty.y;
        }
    }

    grnn_reduce_store(num, den, s_num, s_den, tid, wave, lo, hi, mBase, B, out);
}

// ---------------------------------------------------------------------------
// Fallback path (no workspace): fuse fp32 load + convert + t_sq per tile.
// ---------------------------------------------------------------------------
__global__ __launch_bounds__(256)
void grnn_wmma_fused_kernel(const float* __restrict__ x,
                            const float* __restrict__ Xt,
                            const float* __restrict__ y,
                            float* __restrict__ out,
                            int B, int N) {
    __shared__ float s_num[8][16];
    __shared__ float s_den[8][16];

    const int tid  = threadIdx.x;
    const int lane = tid & 31;
    const int wave = tid >> 5;
    const int lo   = lane & 15;
    const int hi   = lane >> 4;
    const int mBase = blockIdx.x * 16;

    int arow = mBase + lo;
    if (arow >= B) arow = B - 1;
    v16h a0, a1;
    float xs_part = load_frags_f32(x + (size_t)arow * 64, hi, a0, a1);
    float xsq = xs_part + __shfl_xor(xs_part, 16);

    float xs2[8];
#pragma unroll
    for (int r = 0; r < 8; ++r)
        xs2[r] = __shfl(xsq, hi * 8 + r) * -K_HALF_LOG2E;

    float num[8], den[8];
#pragma unroll
    for (int r = 0; r < 8; ++r) { num[r] = 0.0f; den[r] = 0.0f; }

    const int ntiles = (N + 15) >> 4;
    for (int ct = wave; ct < ntiles; ct += 8) {
        const int col  = ct * 16 + lo;
        const int colc = (col < N) ? col : (N - 1);

        v16h b0, b1;
        float ts_part = load_frags_f32(Xt + (size_t)colc * 64, hi, b0, b1);
        float tsq2 = (ts_part + __shfl_xor(ts_part, 16)) * -K_HALF_LOG2E;
        float yv   = y[colc];

        v8f c = {0.f, 0.f, 0.f, 0.f, 0.f, 0.f, 0.f, 0.f};
        c = __builtin_amdgcn_wmma_f32_16x16x32_f16(false, a0, false, b0,
                                                   (short)0, c, false, false);
        c = __builtin_amdgcn_wmma_f32_16x16x32_f16(false, a1, false, b1,
                                                   (short)0, c, false, false);

        const float colmask = (col < N) ? 1.0f : 0.0f;
#pragma unroll
        for (int r = 0; r < 8; ++r) {
            float w = __builtin_amdgcn_exp2f(fmaf(c[r], TWO_K, xs2[r] + tsq2)) * colmask;
            den[r] += w;
            num[r] += w * yv;
        }
    }

    grnn_reduce_store(num, den, s_num, s_den, tid, wave, lo, hi, mBase, B, out);
}

extern "C" void kernel_launch(void* const* d_in, const int* in_sizes, int n_in,
                              void* d_out, int out_size, void* d_ws, size_t ws_size,
                              hipStream_t stream) {
    const float* x  = (const float*)d_in[0];   // [B, 64]
    const float* Xt = (const float*)d_in[1];   // [N, 64]
    const float* y  = (const float*)d_in[2];   // [N]
    float* out = (float*)d_out;                // [B]

    const int F = 64;
    const int B = in_sizes[0] / F;             // 4096
    const int N = in_sizes[2];                 // 100000

    const int grid = (B + 15) / 16;            // one 16-row M tile per block

    // Workspace layout: [ f16 X_train : N*64*2 B ][ float2 {-K*t^2, y} : N*8 B ]
    const size_t xh_bytes = (size_t)N * F * sizeof(_Float16);
    const size_t need = xh_bytes + (size_t)N * sizeof(float2);

    if (ws_size >= need) {
        _Float16* Xh = (_Float16*)d_ws;
        float2*   ty = (float2*)((char*)d_ws + xh_bytes);
        grnn_prep_kernel<<<(N + 255) / 256, 256, 0, stream>>>(Xt, y, Xh, ty, N);
        grnn_wmma_fast_kernel<<<grid, 256, 0, stream>>>(x, Xh, ty, out, B, N);
    } else {
        grnn_wmma_fused_kernel<<<grid, 256, 0, stream>>>(x, Xt, y, out, B, N);
    }
}
